// TransformerAgent_773094113706
// MI455X (gfx1250) — compile-verified, matrix operand order checked by reading, not run on hardware
//
#include <hip/hip_runtime.h>
#include <hip/hip_bf16.h>
#include <math.h>

// ---------------------------------------------------------------------------
// GPT-2 forward (B=4, S=1200, E=512, H=8, HD=64, L=4) for gfx1250 (MI455X).
// All matmuls on v_wmma_f32_16x16x32_f16. Weights, attention V, and the
// structured mask are pre-packed into WMMA-fragment-shaped layouts so the
// hot loops issue only contiguous vector loads + WMMA.
// ---------------------------------------------------------------------------

typedef __attribute__((ext_vector_type(16))) _Float16 v16h;
typedef __attribute__((ext_vector_type(8)))  _Float16 v8h;
typedef __attribute__((ext_vector_type(8)))  float    v8f;

#define E_DIM   512
#define S_LEN   1200
#define B_SZ    4
#define NHEAD   8
#define HDIM    64
#define NLAYER  4
#define MTOT    (B_SZ * S_LEN)        // 4800 rows
#define KT_S    38                    // ceil(S_LEN / 32), zero-padded key tiles
#define T16     (S_LEN / 16)          // 75 16-wide tiles along S

// K index for half h (0..15) of a 16x16x32 f16 fragment, given lane.
__device__ __forceinline__ int kmap(int h, int lane) {
  int grp = (lane >> 4) << 3;        // lanes 16-31 shift K by 8
  return (h < 8 ? h : h + 8) + grp;
}

// A fragment from row-major f16 [rows x lda]: per-lane row = m0 + (lane&15),
// halves sweep K. Two aligned 16B vector loads per lane.
__device__ __forceinline__ v16h load_a_f16(const _Float16* __restrict__ A,
                                           int lda, int m0, int k0, int lane) {
  const _Float16* row =
      A + (size_t)(m0 + (lane & 15)) * lda + k0 + ((lane >> 4) << 3);
  v8h lo = *(const v8h*)(row);
  v8h hi = *(const v8h*)(row + 16);
  return __builtin_shufflevector(lo, hi, 0, 1, 2, 3, 4, 5, 6, 7, 8, 9, 10, 11,
                                 12, 13, 14, 15);
}

// B fragment from pre-packed buffer: one contiguous 32B load per lane.
__device__ __forceinline__ v16h load_b_pk(const _Float16* __restrict__ Wp,
                                          int KT, int kt, int nt, int lane) {
  return *(const v16h*)(Wp + ((((size_t)nt * KT + kt) * 32 + lane) << 4));
}

// ---------------------------------------------------------------------------
// Pack f32 weights [L][K][N] -> f16 B-fragment layout:
//   Wp[((nt*(K/32)+kt)*32 + lane)*16 + h] = W[kt*32+kmap(h,lane)][nt*16+(lane&15)]
// ---------------------------------------------------------------------------
__global__ __launch_bounds__(256)
void pack_w_kernel(const float* __restrict__ W, _Float16* __restrict__ Wp,
                   int K, int N) {
  size_t tid = (size_t)blockIdx.x * 256 + threadIdx.x;    // over K*N
  int l = blockIdx.y;
  const float* Wl = W + (size_t)l * K * N;
  _Float16* Wpl = Wp + (size_t)l * K * N;
  int h    = (int)(tid & 15);
  int lane = (int)((tid >> 4) & 31);
  size_t rest = tid >> 9;
  int kt = (int)(rest % (size_t)(K >> 5));
  int nt = (int)(rest / (size_t)(K >> 5));
  int k = kt * 32 + kmap(h, lane);
  int n = nt * 16 + (lane & 15);
  Wpl[tid] = (_Float16)Wl[(size_t)k * N + n];
}

// ---------------------------------------------------------------------------
// Pack the V part of qkv into B-fragment layout per (b, head), key tiles
// padded to KT_S (out-of-range keys -> 0).
// ---------------------------------------------------------------------------
__global__ __launch_bounds__(256)
void pack_v_kernel(const _Float16* __restrict__ qkv,
                   _Float16* __restrict__ Vp) {
  size_t tid = (size_t)blockIdx.x * 256 + threadIdx.x;
  int h    = (int)(tid & 15);
  int lane = (int)((tid >> 4) & 31);
  size_t rest = tid >> 9;            // (bh*KT_S + kt)*4 + nt
  int nt = (int)(rest & 3);
  size_t rest2 = rest >> 2;
  int kt = (int)(rest2 % KT_S);
  int bh = (int)(rest2 / KT_S);
  int b = bh >> 3, head = bh & 7;
  int key = kt * 32 + kmap(h, lane);
  int d   = nt * 16 + (lane & 15);
  _Float16 v = (_Float16)0.f;
  if (key < S_LEN)
    v = qkv[(size_t)(b * S_LEN + key) * (3 * E_DIM) + 2 * E_DIM +
            head * HDIM + d];
  Vp[tid] = v;
}

// ---------------------------------------------------------------------------
// Pack the bool mask [S][S] into an additive f16 bias tiled to match the
// score C-fragment layout: mb[((qt*T16 + kt)*32 + lane)*8 + r] covers
// qrow = qt*16 + (lane>>4)*8 + r, kcol = kt*16 + (lane&15).
// True -> 0, False -> -60000 (drives exp() to 0 in f32 softmax).
// ---------------------------------------------------------------------------
__global__ __launch_bounds__(256)
void pack_mask_kernel(const unsigned char* __restrict__ mask,
                      _Float16* __restrict__ mb) {
  size_t tid = (size_t)blockIdx.x * 256 + threadIdx.x;    // over S*S
  int r    = (int)(tid & 7);
  int lane = (int)((tid >> 3) & 31);
  size_t rest = tid >> 8;
  int kt = (int)(rest % T16);
  int qt = (int)(rest / T16);
  int qrow = qt * 16 + ((lane >> 4) << 3) + r;
  int kcol = kt * 16 + (lane & 15);
  mb[tid] = mask[(size_t)qrow * S_LEN + kcol] ? (_Float16)0.f
                                              : (_Float16)(-60000.f);
}

// ---------------------------------------------------------------------------
// GEMM: C[M,N] = epilogue(A[M,K](f16) @ Wp(packed f16) + bias [+ residual])
// EPI: 0 = bias, 1 = bias + residual add, 2 = bias + gelu_new
// OUTH: 1 -> f16 output, 0 -> f32 output.
// Block = 128 threads (4 waves); block tile 64x128; wave tile 32x64.
// ---------------------------------------------------------------------------
template <int EPI, int OUTH>
__global__ __launch_bounds__(128)
void gemm_wmma_kernel(const _Float16* __restrict__ A,
                      const _Float16* __restrict__ Wp,
                      const float* __restrict__ bias,
                      const float* __restrict__ res,
                      void* __restrict__ Cout, int M, int N, int K) {
  int lane = threadIdx.x & 31;
  int wave = threadIdx.x >> 5;
  int m0 = blockIdx.y * 64 + (wave & 1) * 32;
  int n0 = blockIdx.x * 128 + (wave >> 1) * 64;
  int KT = K >> 5;

  v8f acc[2][4] = {};
  for (int kt = 0; kt < KT; ++kt) {
    if (kt + 1 < KT) {
      // Next k-step: A rows and the (sequential) packed-B stream.
      __builtin_prefetch(A + (size_t)(m0 + (lane & 15)) * K + (kt + 1) * 32, 0, 3);
      __builtin_prefetch(
          Wp + ((((size_t)(n0 >> 4) * KT + kt + 1) * 32 + lane) << 4), 0, 3);
    }
    v16h a0 = load_a_f16(A, K, m0, kt * 32, lane);
    v16h a1 = load_a_f16(A, K, m0 + 16, kt * 32, lane);
#pragma unroll
    for (int t = 0; t < 4; ++t) {
      v16h b = load_b_pk(Wp, KT, kt, (n0 >> 4) + t, lane);
      acc[0][t] = __builtin_amdgcn_wmma_f32_16x16x32_f16(
          false, a0, false, b, (short)0, acc[0][t], false, false);
      acc[1][t] = __builtin_amdgcn_wmma_f32_16x16x32_f16(
          false, a1, false, b, (short)0, acc[1][t], false, false);
    }
  }

  int grp = (lane >> 4) << 3;
  int cb  = n0 + (lane & 15);
#pragma unroll
  for (int mi = 0; mi < 2; ++mi) {
    int rbase = m0 + mi * 16 + grp;
#pragma unroll
    for (int t = 0; t < 4; ++t) {
      int n = cb + t * 16;
      float bv = bias[n];
#pragma unroll
      for (int r = 0; r < 8; ++r) {
        size_t idx = (size_t)(rbase + r) * N + n;
        float v = acc[mi][t][r] + bv;
        if (EPI == 1) v += res[idx];
        if (EPI == 2) {
          float x = v;
          v = 0.5f * x * (1.0f + tanhf(0.7978845608028654f *
                                       (x + 0.044715f * x * x * x)));
        }
        if (OUTH) ((_Float16*)Cout)[idx] = (_Float16)v;
        else      ((float*)Cout)[idx]    = v;
      }
    }
  }
}

// ---------------------------------------------------------------------------
// LayerNorm over E=512: one wave per row, 16 elements per lane.
// ---------------------------------------------------------------------------
template <typename OUT>
__global__ __launch_bounds__(32)
void ln_kernel(const float* __restrict__ X, const float* __restrict__ g,
               const float* __restrict__ b, OUT* __restrict__ Y) {
  int row  = blockIdx.x;
  int lane = threadIdx.x;
  const float* x = X + (size_t)row * E_DIM;
  float v[16];
  float s = 0.f;
#pragma unroll
  for (int i = 0; i < 16; ++i) { v[i] = x[lane + i * 32]; s += v[i]; }
#pragma unroll
  for (int off = 16; off > 0; off >>= 1) s += __shfl_xor(s, off, 32);
  float mu = s * (1.0f / E_DIM);
  float vs = 0.f;
#pragma unroll
  for (int i = 0; i < 16; ++i) { float d = v[i] - mu; vs += d * d; }
#pragma unroll
  for (int off = 16; off > 0; off >>= 1) vs += __shfl_xor(vs, off, 32);
  float rstd = rsqrtf(vs * (1.0f / E_DIM) + 1e-5f);
  OUT* y = Y + (size_t)row * E_DIM;
#pragma unroll
  for (int i = 0; i < 16; ++i) {
    int c = lane + i * 32;
    y[c] = (OUT)((v[i] - mu) * rstd * g[c] + b[c]);
  }
}

// h = inputs_embeds + wpe (broadcast over batch)
__global__ __launch_bounds__(256)
void embed_kernel(const float* __restrict__ emb, const float* __restrict__ wpe,
                  float* __restrict__ h) {
  int idx = blockIdx.x * 256 + threadIdx.x;        // over MTOT*E_DIM
  int col = idx & (E_DIM - 1);
  int row = idx >> 9;                              // E_DIM = 512 = 2^9
  int s   = row % S_LEN;
  h[idx] = emb[idx] + wpe[(size_t)s * E_DIM + col];
}

// ---------------------------------------------------------------------------
// Fused attention: one wave per (b, head, 16-query tile). Flash-style online
// softmax over 32-key tiles; scores and P*V via WMMA on f16 Q/K and packed V.
// Mask applied as a pre-packed additive f16 bias in score-fragment layout.
// P re-shaped C-layout -> A-layout through LDS.
// ---------------------------------------------------------------------------
__global__ __launch_bounds__(32)
void attn_kernel(const _Float16* __restrict__ qkv,
                 const _Float16* __restrict__ vpack,
                 const _Float16* __restrict__ mbias,
                 _Float16* __restrict__ ctx) {
  __shared__ float Pt[16][32];
  int lane = threadIdx.x;
  int qt   = blockIdx.x;
  int q0   = qt * 16;
  int head = blockIdx.y;
  int b    = blockIdx.z;
  const int ld = 3 * E_DIM;
  const _Float16* Q  = qkv + (size_t)b * S_LEN * ld + head * HDIM;
  const _Float16* Kp = Q + E_DIM;
  const _Float16* Vpk =
      vpack + (size_t)(b * NHEAD + head) * KT_S * 4 * 512;   // 512 = 32*16

  // Q fragments (HD=64 -> two 32-wide K steps)
  v16h qf[2];
#pragma unroll
  for (int s = 0; s < 2; ++s) qf[s] = load_a_f16(Q, ld, q0, s * 32, lane);

  v8f o[4] = {};
  float run_max[8], run_sum[8];
#pragma unroll
  for (int r = 0; r < 8; ++r) { run_max[r] = -1e30f; run_sum[r] = 0.f; }

  int grp = (lane >> 4) << 3;    // row group offset (0 or 8)
  int col = lane & 15;           // column within 16-wide tile

  for (int kb = 0; kb < S_LEN; kb += 32) {
    float p[2][8];
    float tmax[8];
#pragma unroll
    for (int r = 0; r < 8; ++r) tmax[r] = -1e30f;

#pragma unroll
    for (int j = 0; j < 2; ++j) {
      int kc0 = kb + j * 16;
      if (kc0 < S_LEN) {
        v8f sc = {};
#pragma unroll
        for (int s = 0; s < 2; ++s) {
          // B = K^T: per-lane column = key, halves sweep d -> same pattern as
          // an A fragment over K[key, d].
          v16h kf = load_a_f16(Kp, ld, kc0, s * 32, lane);
          sc = __builtin_amdgcn_wmma_f32_16x16x32_f16(
              false, qf[s], false, kf, (short)0, sc, false, false);
        }
        // Mask bias: one 16B vector load, layout matches the C-fragment.
        int kt16 = (kb >> 4) + j;
        v8h mb = *(const v8h*)(
            mbias + ((((size_t)qt * T16 + kt16) * 32 + lane) << 3));
#pragma unroll
        for (int r = 0; r < 8; ++r) {
          float sv = sc[r] * 0.125f + (float)mb[r];   // 1/sqrt(HD) + mask
          p[j][r] = sv;
          tmax[r] = fmaxf(tmax[r], sv);
        }
      } else {
#pragma unroll
        for (int r = 0; r < 8; ++r) p[j][r] = -1e30f;  // exp -> 0
      }
    }

    // Online softmax update (each row lives across a 16-lane half-wave).
#pragma unroll
    for (int r = 0; r < 8; ++r) {
      float m = tmax[r];
#pragma unroll
      for (int off = 8; off > 0; off >>= 1) m = fmaxf(m, __shfl_xor(m, off, 16));
      float nm    = fmaxf(run_max[r], m);
      float alpha = __expf(run_max[r] - nm);
      run_max[r]  = nm;
      float lsum = 0.f;
#pragma unroll
      for (int j = 0; j < 2; ++j) {
        float pv = __expf(p[j][r] - nm);
        p[j][r] = pv;
        lsum += pv;
      }
#pragma unroll
      for (int off = 8; off > 0; off >>= 1) lsum += __shfl_xor(lsum, off, 16);
      run_sum[r] = run_sum[r] * alpha + lsum;
#pragma unroll
      for (int t = 0; t < 4; ++t) o[t][r] *= alpha;
    }

    // C-layout -> A-layout through LDS.
#pragma unroll
    for (int j = 0; j < 2; ++j)
#pragma unroll
      for (int r = 0; r < 8; ++r) Pt[grp + r][j * 16 + col] = p[j][r];
    __syncthreads();
    v16h pf;
#pragma unroll
    for (int h = 0; h < 16; ++h)
      pf[h] = (_Float16)Pt[lane & 15][kmap(h, lane)];
    __syncthreads();

    // O += P @ V : packed V fragment = one contiguous 32B load per lane.
    int ktile = kb >> 5;
#pragma unroll
    for (int t = 0; t < 4; ++t) {
      v16h vf = *(const v16h*)(
          Vpk + ((((size_t)ktile * 4 + t) * 32 + lane) << 4));
      o[t] = __builtin_amdgcn_wmma_f32_16x16x32_f16(
          false, pf, false, vf, (short)0, o[t], false, false);
    }
  }

  // Normalize and scatter back to [B,S,E] f16 (heads interleaved).
  _Float16* out =
      ctx + ((size_t)b * S_LEN + q0 + grp) * E_DIM + head * HDIM + col;
#pragma unroll
  for (int r = 0; r < 8; ++r) {
    float inv = 1.0f / run_sum[r];
#pragma unroll
    for (int t = 0; t < 4; ++t)
      out[(size_t)r * E_DIM + t * 16] = (_Float16)(o[t][r] * inv);
  }
}

// ---------------------------------------------------------------------------
extern "C" void kernel_launch(void* const* d_in, const int* in_sizes, int n_in,
                              void* d_out, int out_size, void* d_ws,
                              size_t ws_size, hipStream_t stream) {
  (void)in_sizes; (void)n_in; (void)out_size; (void)ws_size;
  const float* emb    = (const float*)d_in[0];
  const float* wpe    = (const float*)d_in[1];
  const unsigned char* mask = (const unsigned char*)d_in[2];  // numpy bool
  const float* ln1_g  = (const float*)d_in[3];
  const float* ln1_b  = (const float*)d_in[4];
  const float* attn_w = (const float*)d_in[5];
  const float* attn_b = (const float*)d_in[6];
  const float* proj_w = (const float*)d_in[7];
  const float* proj_b = (const float*)d_in[8];
  const float* ln2_g  = (const float*)d_in[9];
  const float* ln2_b  = (const float*)d_in[10];
  const float* fc_w   = (const float*)d_in[11];
  const float* fc_b   = (const float*)d_in[12];
  const float* fc2_w  = (const float*)d_in[13];
  const float* fc2_b  = (const float*)d_in[14];
  const float* lnf_g  = (const float*)d_in[15];
  const float* lnf_b  = (const float*)d_in[16];

  const int E  = E_DIM;
  const int E3 = 3 * E_DIM;
  const int E4 = 4 * E_DIM;
  const size_t ME = (size_t)MTOT * E_DIM;
  const size_t VP = (size_t)B_SZ * NHEAD * KT_S * 4 * 512;  // packed V halves
  const size_t MB = (size_t)S_LEN * S_LEN;                  // packed mask halves

  // Workspace carve-up (all region sizes are multiples of 256B).
  char* ws = (char*)d_ws;
  float*     h      = (float*)ws;      ws += ME * 4;            // [M,E]  f32
  _Float16*  x_h    = (_Float16*)ws;   ws += ME * 2;            // [M,E]  f16
  _Float16*  qkv_h  = (_Float16*)ws;   ws += 3 * ME * 2;        // [M,3E] f16
  _Float16*  ctx_h  = (_Float16*)ws;   ws += ME * 2;            // [M,E]  f16
  _Float16*  mlp_h  = (_Float16*)ws;   ws += 4 * ME * 2;        // [M,4E] f16
  _Float16*  vpk    = (_Float16*)ws;   ws += VP * 2;            // packed V
  _Float16*  mbias  = (_Float16*)ws;   ws += MB * 2;            // packed mask
  _Float16*  pk_attn = (_Float16*)ws;  ws += (size_t)NLAYER * E * E3 * 2;
  _Float16*  pk_proj = (_Float16*)ws;  ws += (size_t)NLAYER * E * E  * 2;
  _Float16*  pk_fc   = (_Float16*)ws;  ws += (size_t)NLAYER * E * E4 * 2;
  _Float16*  pk_fc2  = (_Float16*)ws;  ws += (size_t)NLAYER * E4 * E * 2;

  // One-time packing: weights -> B-fragment layout, mask -> bias tiles.
  pack_w_kernel<<<dim3((E * E3) / 256, NLAYER), 256, 0, stream>>>(attn_w, pk_attn, E, E3);
  pack_w_kernel<<<dim3((E * E)  / 256, NLAYER), 256, 0, stream>>>(proj_w, pk_proj, E, E);
  pack_w_kernel<<<dim3((E * E4) / 256, NLAYER), 256, 0, stream>>>(fc_w,   pk_fc,   E, E4);
  pack_w_kernel<<<dim3((E4 * E) / 256, NLAYER), 256, 0, stream>>>(fc2_w,  pk_fc2,  E4, E);
  pack_mask_kernel<<<(unsigned)(MB / 256), 256, 0, stream>>>(mask, mbias);

  embed_kernel<<<(MTOT * E) / 256, 256, 0, stream>>>(emb, wpe, h);

  for (int l = 0; l < NLAYER; ++l) {
    // --- attention block ---
    ln_kernel<_Float16><<<MTOT, 32, 0, stream>>>(h, ln1_g + l * E, ln1_b + l * E, x_h);
    gemm_wmma_kernel<0, 1><<<dim3(E3 / 128, MTOT / 64), 128, 0, stream>>>(
        x_h, pk_attn + (size_t)l * E * E3, attn_b + (size_t)l * E3, nullptr,
        qkv_h, MTOT, E3, E);
    pack_v_kernel<<<(unsigned)(VP / 256), 256, 0, stream>>>(qkv_h, vpk);
    attn_kernel<<<dim3(S_LEN / 16, NHEAD, B_SZ), 32, 0, stream>>>(
        qkv_h, vpk, mbias, ctx_h);
    gemm_wmma_kernel<1, 0><<<dim3(E / 128, MTOT / 64), 128, 0, stream>>>(
        ctx_h, pk_proj + (size_t)l * E * E, proj_b + (size_t)l * E, h, h,
        MTOT, E, E);
    // --- MLP block ---
    ln_kernel<_Float16><<<MTOT, 32, 0, stream>>>(h, ln2_g + l * E, ln2_b + l * E, x_h);
    gemm_wmma_kernel<2, 1><<<dim3(E4 / 128, MTOT / 64), 128, 0, stream>>>(
        x_h, pk_fc + (size_t)l * E * E4, fc_b + (size_t)l * E4, nullptr,
        mlp_h, MTOT, E4, E);
    gemm_wmma_kernel<1, 0><<<dim3(E / 128, MTOT / 64), 128, 0, stream>>>(
        mlp_h, pk_fc2 + (size_t)l * E4 * E, fc2_b + (size_t)l * E, h, h,
        MTOT, E, E4);
  }

  ln_kernel<float><<<MTOT, 32, 0, stream>>>(h, lnf_g, lnf_b, (float*)d_out);
}
